// Net_74560632259166
// MI455X (gfx1250) — matrix operand order repaired and checked by reading, not compiled, not run on hardware
//
#include <hip/hip_runtime.h>

// ---------------------------------------------------------------------------
// CDNA5 (gfx1250) Point Transformer forward.
// Dense linear layers run on v_wmma_f32_16x16x32_bf16 (wave32 WMMA),
// 4 WMMAs per wave per K-step (32x32 accumulator), vectorized LDS staging.
// ---------------------------------------------------------------------------

typedef __attribute__((ext_vector_type(16))) __bf16 v16bf;
typedef __attribute__((ext_vector_type(8)))  float  v8f;
typedef __attribute__((ext_vector_type(8)))  short  v8s;

__device__ __forceinline__ unsigned short f2bf(float f) {
  unsigned u = __float_as_uint(f);
  u += 0x7FFFu + ((u >> 16) & 1u);           // round-to-nearest-even
  return (unsigned short)(u >> 16);
}

// ======================= WMMA GEMM: C[M,N] = A[M,K] * W[N,K]^T (+bias)(ReLU)
#define TM 128
#define TN 64
#define GK 32
#define LDA_S 40   // padded LDS stride (shorts); 80B rows => 16B aligned chunks
#define LDB_S 40

__global__ __launch_bounds__(256)
void gemm_bf16_kernel(const float* __restrict__ A, const float* __restrict__ W,
                      const float* __restrict__ bias, float* __restrict__ C,
                      int M, int N, int K, int act)
{
  __shared__ unsigned short sA[TM * LDA_S];
  __shared__ unsigned short sB[TN * LDB_S];

  const int tid  = threadIdx.x;
  const int lane = tid & 31;
  const int wave = tid >> 5;      // 0..7
  const int wm   = wave >> 1;     // 0..3 : 32-row band
  const int wn   = wave & 1;      // 0..1 : 32-col band
  const int m0   = blockIdx.y * TM;
  const int n0   = blockIdx.x * TN;

  const int half = lane >> 4;     // ISA fragment lane-group
  const int row  = lane & 15;

  const bool kvec = ((K & 3) == 0);

  v8f acc[2][2] = {};

  for (int k0 = 0; k0 < K; k0 += GK) {
    // ---------------- stage A tile (TM x GK) ----------------
    if (kvec && (m0 + TM <= M)) {
#pragma unroll
      for (int it = 0; it < (TM * GK) / (256 * 4); ++it) {   // 4 iters
        int e = (tid + it * 256) * 4;
        int r = e >> 5, c = e & 31;
        const float* src = A + (size_t)(m0 + r) * K + (k0 + c);
        float4 v4 = {0.f, 0.f, 0.f, 0.f};
        if (k0 + c + 4 <= K) {
          v4 = *(const float4*)src;
          if (k0 + c + GK + 4 <= K) __builtin_prefetch(src + GK, 0, 1);
        }
        union { unsigned short h[4]; unsigned long long u; } pk;
        pk.h[0] = f2bf(v4.x); pk.h[1] = f2bf(v4.y);
        pk.h[2] = f2bf(v4.z); pk.h[3] = f2bf(v4.w);
        *(unsigned long long*)&sA[r * LDA_S + c] = pk.u;
      }
    } else {
      for (int e = tid; e < TM * GK; e += 256) {
        int r = e >> 5, c = e & 31;
        int gm = m0 + r, gk = k0 + c;
        float v = (gm < M && gk < K) ? A[(size_t)gm * K + gk] : 0.0f;
        sA[r * LDA_S + c] = f2bf(v);
      }
    }
    // ---------------- stage B tile = W rows (TN x GK) ----------------
    if (kvec && (n0 + TN <= N)) {
#pragma unroll
      for (int it = 0; it < (TN * GK) / (256 * 4); ++it) {   // 2 iters
        int e = (tid + it * 256) * 4;
        int r = e >> 5, c = e & 31;
        float4 v4 = {0.f, 0.f, 0.f, 0.f};
        if (k0 + c + 4 <= K) v4 = *(const float4*)(W + (size_t)(n0 + r) * K + (k0 + c));
        union { unsigned short h[4]; unsigned long long u; } pk;
        pk.h[0] = f2bf(v4.x); pk.h[1] = f2bf(v4.y);
        pk.h[2] = f2bf(v4.z); pk.h[3] = f2bf(v4.w);
        *(unsigned long long*)&sB[r * LDB_S + c] = pk.u;
      }
    } else {
      for (int e = tid; e < TN * GK; e += 256) {
        int r = e >> 5, c = e & 31;
        int gn = n0 + r, gk = k0 + c;
        float v = (gn < N && gk < K) ? W[(size_t)gn * K + gk] : 0.0f;
        sB[r * LDB_S + c] = f2bf(v);
      }
    }
    __syncthreads();

    // ---- A fragments (16x32 bf16): half 0 -> K{0-7,16-23}, half 1 -> K{8-15,24-31}
    v16bf af[2], bf[2];
#pragma unroll
    for (int i = 0; i < 2; ++i) {
      const unsigned short* ar = &sA[(wm * 32 + i * 16 + row) * LDA_S];
      v8s lo = *(const v8s*)(ar + half * 8);
      v8s hi = *(const v8s*)(ar + 16 + half * 8);
      af[i] = __builtin_bit_cast(v16bf, __builtin_shufflevector(lo, hi,
                0,1,2,3,4,5,6,7,8,9,10,11,12,13,14,15));
    }
    // ---- B fragments (32x16 bf16): col = lane&15; half 0 -> K{0-7,16-23}, half 1 -> K{8-15,24-31}
#pragma unroll
    for (int j = 0; j < 2; ++j) {
      const unsigned short* br = &sB[(wn * 32 + j * 16 + row) * LDB_S];
      v8s lo = *(const v8s*)(br + half * 8);
      v8s hi = *(const v8s*)(br + 16 + half * 8);
      bf[j] = __builtin_bit_cast(v16bf, __builtin_shufflevector(lo, hi,
                0,1,2,3,4,5,6,7,8,9,10,11,12,13,14,15));
    }
#pragma unroll
    for (int i = 0; i < 2; ++i)
#pragma unroll
      for (int j = 0; j < 2; ++j)
        acc[i][j] = __builtin_amdgcn_wmma_f32_16x16x32_bf16(
            false, af[i], false, bf[j], (short)0, acc[i][j], false, false);
    __syncthreads();
  }

  // ---- epilogue: C layout: VGPR r -> M = r + 8*half, N = lane&15
#pragma unroll
  for (int j = 0; j < 2; ++j) {
    int nn = n0 + wn * 32 + j * 16 + row;
    if (nn < N) {
      float bvv = bias ? bias[nn] : 0.0f;
#pragma unroll
      for (int i = 0; i < 2; ++i) {
        int mbase = m0 + wm * 32 + i * 16 + 8 * half;
#pragma unroll
        for (int r = 0; r < 8; ++r) {
          int mm = mbase + r;
          if (mm < M) {
            float v = acc[i][j][r] + bvv;
            if (act == 1) v = fmaxf(v, 0.0f);
            C[(size_t)mm * N + nn] = v;
          }
        }
      }
    }
  }
}

// ======================= zero-pad columns (K -> Kpad) ======================
__global__ void pad_cols_kernel(const float* __restrict__ src, float* __restrict__ dst,
                                int M, int Ks, int Kd) {
  int t = blockIdx.x * 256 + threadIdx.x;
  if (t < M * Kd) {
    int c = t % Kd, r = t / Kd;
    dst[t] = (c < Ks) ? src[(size_t)r * Ks + c] : 0.0f;
  }
}

// ======================= BatchNorm (training batch stats) ==================
__global__ void colstats_kernel(const float* __restrict__ x,
                                float* __restrict__ mean, float* __restrict__ rstd,
                                int M, int C) {
  int c = blockIdx.x;
  int tid = threadIdx.x;
  float s = 0.f, s2 = 0.f;
  for (int i = tid; i < M; i += 256) {
    float v = x[(size_t)i * C + c];
    s += v; s2 += v * v;
  }
  __shared__ float rs[256], rq[256];
  rs[tid] = s; rq[tid] = s2;
  __syncthreads();
  for (int st = 128; st > 0; st >>= 1) {
    if (tid < st) { rs[tid] += rs[tid + st]; rq[tid] += rq[tid + st]; }
    __syncthreads();
  }
  if (tid == 0) {
    float m = rs[0] / (float)M;
    float v = rq[0] / (float)M - m * m;
    mean[c] = m;
    rstd[c] = rsqrtf(fmaxf(v, 0.f) + 1e-5f);
  }
}

__global__ void bn_relu_kernel(float* __restrict__ x, const float* __restrict__ mean,
                               const float* __restrict__ rstd, const float* __restrict__ g,
                               const float* __restrict__ beta, int M, int C) {
  int t = blockIdx.x * 256 + threadIdx.x;
  if (t < M * C) {
    int c = t % C;
    float v = (x[t] - mean[c]) * rstd[c] * g[c] + beta[c];
    x[t] = fmaxf(v, 0.f);
  }
}

// ======================= brute-force kNN ===================================
template <int KK>
__global__ __launch_bounds__(256)
void knn_kernel(const float* __restrict__ qry, const float* __restrict__ keys,
                int* __restrict__ idx_out, float* __restrict__ dst_out,
                int Nq, int Nk, int excl) {
  __shared__ float sk[256 * 3];
  int qi = blockIdx.x * 256 + threadIdx.x;
  float qx = 0.f, qy = 0.f, qz = 0.f;
  if (qi < Nq) { qx = qry[qi*3]; qy = qry[qi*3+1]; qz = qry[qi*3+2]; }
  float bd[KK]; int bi[KK];
#pragma unroll
  for (int j = 0; j < KK; ++j) { bd[j] = __builtin_inff(); bi[j] = 0; }
  for (int t0 = 0; t0 < Nk; t0 += 256) {
    int nt = Nk - t0; if (nt > 256) nt = 256;
    __syncthreads();
    if ((int)threadIdx.x < nt) {
      sk[threadIdx.x*3+0] = keys[(size_t)(t0+threadIdx.x)*3+0];
      sk[threadIdx.x*3+1] = keys[(size_t)(t0+threadIdx.x)*3+1];
      sk[threadIdx.x*3+2] = keys[(size_t)(t0+threadIdx.x)*3+2];
    }
    __syncthreads();
    for (int t = 0; t < nt; ++t) {
      int kid = t0 + t;
      float dx = qx - sk[t*3+0], dy = qy - sk[t*3+1], dz = qz - sk[t*3+2];
      float d = dx*dx + dy*dy + dz*dz;
      if (excl && kid == qi) d = __builtin_inff();
      if (d < bd[KK-1]) {
        bd[KK-1] = d; bi[KK-1] = kid;
#pragma unroll
        for (int j = KK-1; j > 0; --j) {
          if (bd[j] < bd[j-1]) {
            float td = bd[j]; bd[j] = bd[j-1]; bd[j-1] = td;
            int ti = bi[j]; bi[j] = bi[j-1]; bi[j-1] = ti;
          }
        }
      }
    }
  }
  if (qi < Nq) {
#pragma unroll
    for (int j = 0; j < KK; ++j) {
      idx_out[(size_t)qi*KK + j] = bi[j];
      if (dst_out) dst_out[(size_t)qi*KK + j] = bd[j];
    }
  }
}

// ======================= farthest point sampling ===========================
__global__ __launch_bounds__(1024)
void fps_kernel(const float* __restrict__ pos, int* __restrict__ out,
                int N, int n_sub) {
  __shared__ float mind[8192];
  __shared__ float rv[1024];
  __shared__ int   ri[1024];
  __shared__ int   s_next;
  int tid = threadIdx.x;
  float x0 = pos[0], y0 = pos[1], z0 = pos[2];
  for (int i = tid; i < N; i += 1024) {
    float dx = pos[i*3]-x0, dy = pos[i*3+1]-y0, dz = pos[i*3+2]-z0;
    mind[i] = dx*dx + dy*dy + dz*dz;
  }
  if (tid == 0) out[0] = 0;
  __syncthreads();
  for (int t = 1; t < n_sub; ++t) {
    float bv = -1.f; int bidx = 0x7fffffff;
    for (int i = tid; i < N; i += 1024) {
      float v = mind[i];
      if (v > bv || (v == bv && i < bidx)) { bv = v; bidx = i; }
    }
    rv[tid] = bv; ri[tid] = bidx;
    __syncthreads();
    for (int st = 512; st > 0; st >>= 1) {
      if (tid < st) {
        float ov = rv[tid+st]; int oi = ri[tid+st];
        if (ov > rv[tid] || (ov == rv[tid] && oi < ri[tid])) { rv[tid]=ov; ri[tid]=oi; }
      }
      __syncthreads();
    }
    if (tid == 0) { s_next = ri[0]; out[t] = ri[0]; }
    __syncthreads();
    int nxt = s_next;
    float nx = pos[nxt*3], ny = pos[nxt*3+1], nz = pos[nxt*3+2];
    for (int i = tid; i < N; i += 1024) {
      float dx = pos[i*3]-nx, dy = pos[i*3+1]-ny, dz = pos[i*3+2]-nz;
      float d = dx*dx + dy*dy + dz*dz;
      if (d < mind[i]) mind[i] = d;
    }
    __syncthreads();
  }
}

// ======================= fused elementwise / gather kernels ================
__global__ void gather3_kernel(const float* __restrict__ pos, const int* __restrict__ idx,
                               float* __restrict__ out, int n) {
  int t = blockIdx.x * 256 + threadIdx.x;
  if (t < n * 3) out[t] = pos[(size_t)idx[t/3] * 3 + (t % 3)];
}

// rel (stride 4, zero-padded 4th component for the K=4 fast-path GEMM)
__global__ void rel_kernel(const float* __restrict__ pos, const int* __restrict__ nbr,
                           float* __restrict__ rel, int N) {
  int t = blockIdx.x * 256 + threadIdx.x;
  if (t < N * 16) {
    int i = t >> 4;
    int nb = nbr[t];
    rel[(size_t)t*4+0] = pos[(size_t)i*3+0] - pos[(size_t)nb*3+0];
    rel[(size_t)t*4+1] = pos[(size_t)i*3+1] - pos[(size_t)nb*3+1];
    rel[(size_t)t*4+2] = pos[(size_t)i*3+2] - pos[(size_t)nb*3+2];
    rel[(size_t)t*4+3] = 0.f;
  }
}

__global__ void alpha_in_kernel(const float* __restrict__ q, const float* __restrict__ kf,
                                const float* __restrict__ delta, const int* __restrict__ nbr,
                                float* __restrict__ out, int N, int C) {
  int t = blockIdx.x * 256 + threadIdx.x;
  int total = N * 16 * C;
  if (t < total) {
    int c  = t % C;
    int ij = t / C;
    int i  = ij >> 4;
    int nb = nbr[ij];
    out[t] = q[(size_t)i*C + c] - kf[(size_t)nb*C + c] + delta[t];
  }
}

// per-channel softmax over 16 neighbors fused with weighted value+delta sum
__global__ void attn_kernel(const float* __restrict__ araw, const float* __restrict__ v,
                            const float* __restrict__ delta, const int* __restrict__ nbr,
                            float* __restrict__ out, int N, int C) {
  int t = blockIdx.x * 256 + threadIdx.x;
  if (t >= N * C) return;
  int c = t % C, i = t / C;
  float a[16];
  float mx = -__builtin_inff();
#pragma unroll
  for (int j = 0; j < 16; ++j) {
    a[j] = araw[((size_t)i*16 + j)*C + c];
    mx = fmaxf(mx, a[j]);
  }
  float s = 0.f;
#pragma unroll
  for (int j = 0; j < 16; ++j) { a[j] = __expf(a[j] - mx); s += a[j]; }
  float inv = 1.f / s, acc = 0.f;
#pragma unroll
  for (int j = 0; j < 16; ++j) {
    int nb = nbr[i*16 + j];
    acc += a[j] * inv * (v[(size_t)nb*C + c] + delta[((size_t)i*16 + j)*C + c]);
  }
  out[t] = acc;
}

__global__ void neigh_max_kernel(const float* __restrict__ x, const int* __restrict__ nbr,
                                 float* __restrict__ out, int Nsub, int C) {
  int t = blockIdx.x * 256 + threadIdx.x;
  if (t >= Nsub * C) return;
  int c = t % C, i = t / C;
  float m = -__builtin_inff();
#pragma unroll
  for (int j = 0; j < 16; ++j)
    m = fmaxf(m, x[(size_t)nbr[i*16 + j]*C + c]);
  out[t] = m;
}

// out = base + knn_interpolate(xsub, idx3, dist3)
__global__ void interp_add_kernel(const float* __restrict__ base, const float* __restrict__ xsub,
                                  const int* __restrict__ idx, const float* __restrict__ dist,
                                  float* __restrict__ out, int N, int C) {
  int t = blockIdx.x * 256 + threadIdx.x;
  if (t >= N * C) return;
  int c = t % C, i = t / C;
  float wsum = 0.f, acc = 0.f;
#pragma unroll
  for (int j = 0; j < 3; ++j) {
    float w = 1.f / fmaxf(dist[i*3 + j], 1e-16f);
    wsum += w;
    acc  += w * xsub[(size_t)idx[i*3 + j]*C + c];
  }
  out[t] = base[t] + acc / wsum;
}

__global__ void logsoftmax_kernel(float* __restrict__ x, int N, int C) {
  int i = blockIdx.x * 256 + threadIdx.x;
  if (i >= N) return;
  float mx = -__builtin_inff();
  for (int c = 0; c < C; ++c) mx = fmaxf(mx, x[(size_t)i*C + c]);
  float s = 0.f;
  for (int c = 0; c < C; ++c) s += expf(x[(size_t)i*C + c] - mx);
  float l = logf(s) + mx;
  for (int c = 0; c < C; ++c) x[(size_t)i*C + c] -= l;
}

// ======================= host-side orchestration ===========================
namespace {

struct Ws { char* p; size_t off, cap; };
static inline void* wmem(Ws& w, size_t bytes) {
  size_t a = (w.off + 255) & ~(size_t)255;
  w.off = a + bytes;
  return w.p + a;
}
static inline float* wf(Ws& w, size_t n) { return (float*)wmem(w, n * 4); }
static inline int*   wi(Ws& w, size_t n) { return (int*)wmem(w, n * 4); }
static inline unsigned cdiv(size_t a, size_t b) { return (unsigned)((a + b - 1) / b); }

struct Lin { const float* w; const float* b; };
struct BNp { const float* w; const float* b; const float* g; const float* beta; };
struct TBp { Lin lin_in; const float* Wq; const float* Wk; const float* Wv;
             Lin pos1, pos2, att1, att2, lin_out; };

static void gemm(hipStream_t s, const float* A, const float* W, const float* b,
                 float* C, int M, int N, int K, int act) {
  dim3 g(cdiv(N, TN), cdiv(M, TM));
  gemm_bf16_kernel<<<g, 256, 0, s>>>(A, W, b, C, M, N, K, act);
}

static void run_knn(hipStream_t s, const float* q, const float* keys,
                    int* idx, float* dist, int Nq, int Nk, int k, int excl) {
  dim3 g(cdiv(Nq, 256));
  if (k == 16) knn_kernel<16><<<g, 256, 0, s>>>(q, keys, idx, dist, Nq, Nk, excl);
  else         knn_kernel<3> <<<g, 256, 0, s>>>(q, keys, idx, dist, Nq, Nk, excl);
}

static void run_mlp_bn(hipStream_t s, Ws& ws, const BNp& p, const float* x,
                       int M, int IC, int OC, float* out) {
  size_t mark = ws.off;
  gemm(s, x, p.w, p.b, out, M, OC, IC, 0);
  float* mean = wf(ws, OC);
  float* rstd = wf(ws, OC);
  colstats_kernel<<<OC, 256, 0, s>>>(out, mean, rstd, M, OC);
  bn_relu_kernel<<<cdiv((size_t)M * OC, 256), 256, 0, s>>>(out, mean, rstd, p.g, p.beta, M, OC);
  ws.off = mark;
}

static void run_tblock(hipStream_t s, Ws& ws, const TBp& p,
                       const float* x_in, int N, int IC, int H, int OC,
                       const float* pos, const int* nbr, float* x_out) {
  size_t mark = ws.off;
  size_t NK = (size_t)N * 16;
  float* x1 = wf(ws, (size_t)N * H);
  gemm(s, x_in, p.lin_in.w, p.lin_in.b, x1, N, H, IC, 1);
  float* q  = wf(ws, (size_t)N * H);
  float* kf = wf(ws, (size_t)N * H);
  float* v  = wf(ws, (size_t)N * H);
  gemm(s, x1, p.Wq, nullptr, q,  N, H, H, 0);
  gemm(s, x1, p.Wk, nullptr, kf, N, H, H, 0);
  gemm(s, x1, p.Wv, nullptr, v,  N, H, H, 0);
  float* rel = wf(ws, NK * 4);                      // K padded 3 -> 4
  rel_kernel<<<cdiv(NK, 256), 256, 0, s>>>(pos, nbr, rel, N);
  float* w4 = wf(ws, 64 * 4);                       // pos1.w padded [64,3] -> [64,4]
  pad_cols_kernel<<<cdiv(64 * 4, 256), 256, 0, s>>>(p.pos1.w, w4, 64, 3, 4);
  float* hid = wf(ws, NK * 64);
  gemm(s, rel, w4, p.pos1.b, hid, (int)NK, 64, 4, 1);
  float* delta = wf(ws, NK * H);
  gemm(s, hid, p.pos2.w, p.pos2.b, delta, (int)NK, H, 64, 0);
  float* ain = wf(ws, NK * H);
  alpha_in_kernel<<<cdiv(NK * H, 256), 256, 0, s>>>(q, kf, delta, nbr, ain, N, H);
  gemm(s, ain, p.att1.w, p.att1.b, hid, (int)NK, 64, H, 1);   // reuse hid
  gemm(s, hid, p.att2.w, p.att2.b, ain, (int)NK, H, 64, 0);   // reuse ain as alpha_raw
  attn_kernel<<<cdiv((size_t)N * H, 256), 256, 0, s>>>(ain, v, delta, nbr, q, N, H); // reuse q
  gemm(s, q, p.lin_out.w, p.lin_out.b, x_out, N, OC, H, 1);
  ws.off = mark;
}

} // namespace

extern "C" void kernel_launch(void* const* d_in, const int* in_sizes, int n_in,
                              void* d_out, int out_size, void* d_ws, size_t ws_size,
                              hipStream_t stream) {
  (void)in_sizes; (void)n_in; (void)out_size;
  const int Ns[5]   = {8192, 2048, 512, 128, 32};
  const int DIMS[5] = {32, 64, 128, 256, 512};

  const float* x_in = (const float*)d_in[0];
  const float* pos0 = (const float*)d_in[1];

  // ---- parameter cursor: recursive insertion-order flatten of params dict ----
  int cur = 2;
  auto nx  = [&]() -> const float* { return (const float*)d_in[cur++]; };
  auto lin = [&](bool bias) { Lin L; L.w = nx(); L.b = bias ? nx() : nullptr; return L; };
  auto bnp = [&]() { BNp B; B.w = nx(); B.b = nx(); B.g = nx(); B.beta = nx(); return B; };
  auto tbp = [&]() {
    TBp t; t.lin_in = lin(true);
    t.Wq = nx(); t.Wk = nx(); t.Wv = nx();
    t.pos1 = lin(true); t.pos2 = lin(true);
    t.att1 = lin(true); t.att2 = lin(true);
    t.lin_out = lin(true); return t;
  };

  BNp mlp_input = bnp();
  TBp tdown[4];      for (int i = 0; i < 4; ++i) tdown[i]      = tbp();
  BNp mlp_down[4];   for (int i = 0; i < 4; ++i) mlp_down[i]   = bnp();
  BNp mlp_up_sub[4]; for (int i = 0; i < 4; ++i) mlp_up_sub[i] = bnp();
  BNp mlp_up[4];     for (int i = 0; i < 4; ++i) mlp_up[i]     = bnp();
  TBp tup[4];        for (int i = 0; i < 4; ++i) tup[i]        = tbp();
  TBp summit_t = tbp();
  BNp mlp_summit = bnp();
  TBp end_t = tbp();
  Lin head1 = lin(true), head2 = lin(true), head3 = lin(true);

  // ---- persistent workspace region ----
  Ws ws{(char*)d_ws, 0, ws_size};
  float* xs[5];
  for (int i = 0; i < 5; ++i) xs[i] = wf(ws, (size_t)Ns[i] * DIMS[i]);
  const float* poss[5];
  poss[0] = pos0;
  float* posbuf[4];
  for (int i = 0; i < 4; ++i) { posbuf[i] = wf(ws, (size_t)Ns[i+1] * 3); poss[i+1] = posbuf[i]; }
  float* xA = wf(ws, (size_t)8192 * 64);
  float* xB = wf(ws, (size_t)8192 * 64);

  // ---- input MLP + BN + ReLU (pad K: 6 -> 8 for the vectorized fast path) ----
  {
    size_t mark = ws.off;
    float* xpad = wf(ws, (size_t)8192 * 8);
    pad_cols_kernel<<<cdiv((size_t)8192 * 8, 256), 256, 0, stream>>>(x_in, xpad, 8192, 6, 8);
    float* wpad = wf(ws, 32 * 8);
    pad_cols_kernel<<<cdiv(32 * 8, 256), 256, 0, stream>>>(mlp_input.w, wpad, 32, 6, 8);
    gemm(stream, xpad, wpad, mlp_input.b, xs[0], 8192, 32, 8, 0);
    float* mean = wf(ws, 32);
    float* rstd = wf(ws, 32);
    colstats_kernel<<<32, 256, 0, stream>>>(xs[0], mean, rstd, 8192, 32);
    bn_relu_kernel<<<cdiv((size_t)8192 * 32, 256), 256, 0, stream>>>(
        xs[0], mean, rstd, mlp_input.g, mlp_input.beta, 8192, 32);
    ws.off = mark;
  }

  // ---- encoder: tblock -> FPS -> knn -> mlp -> neighborhood max ----
  for (int i = 0; i < 4; ++i) {
    size_t mark = ws.off;
    int Ni = Ns[i], H = DIMS[i], OC = DIMS[i+1], Nsub = Ns[i+1];
    int* nbr = wi(ws, (size_t)Ni * 16);
    run_knn(stream, poss[i], poss[i], nbr, nullptr, Ni, Ni, 16, 1);
    float* xt = wf(ws, (size_t)Ni * OC);
    run_tblock(stream, ws, tdown[i], xs[i], Ni, H, H, OC, poss[i], nbr, xt);
    int* sub = wi(ws, Nsub);
    fps_kernel<<<1, 1024, 0, stream>>>(poss[i], sub, Ni, Nsub);
    gather3_kernel<<<cdiv((size_t)Nsub * 3, 256), 256, 0, stream>>>(poss[i], sub, posbuf[i], Nsub);
    int* nbr2 = wi(ws, (size_t)Nsub * 16);
    run_knn(stream, poss[i+1], poss[i], nbr2, nullptr, Nsub, Ni, 16, 0);
    float* xm = wf(ws, (size_t)Ni * OC);
    run_mlp_bn(stream, ws, mlp_down[i], xt, Ni, OC, OC, xm);
    neigh_max_kernel<<<cdiv((size_t)Nsub * OC, 256), 256, 0, stream>>>(xm, nbr2, xs[i+1], Nsub, OC);
    ws.off = mark;
  }

  // ---- summit ----
  {
    size_t mark = ws.off;
    int* nbr = wi(ws, 32 * 16);
    run_knn(stream, poss[4], poss[4], nbr, nullptr, 32, 32, 16, 1);
    float* xt = wf(ws, (size_t)32 * 512);
    run_tblock(stream, ws, summit_t, xs[4], 32, 512, 512, 512, poss[4], nbr, xt);
    run_mlp_bn(stream, ws, mlp_summit, xt, 32, 512, 512, xA);
    ws.off = mark;
  }

  // ---- decoder: tblock -> mlp_sub -> knn_interpolate + skip-mlp add ----
  float* xcur = xA;
  float* xalt = xB;
  for (int i = 0; i < 4; ++i) {
    int j = 3 - i;
    int Ncur = Ns[j+1], Nup = Ns[j];
    int ICt = DIMS[j+1], Ht = DIMS[j+1], OCt = DIMS[j];
    size_t mark = ws.off;
    int* nbr = wi(ws, (size_t)Ncur * 16);
    run_knn(stream, poss[j+1], poss[j+1], nbr, nullptr, Ncur, Ncur, 16, 1);
    float* xt = wf(ws, (size_t)Ncur * OCt);
    run_tblock(stream, ws, tup[j], xcur, Ncur, ICt, Ht, OCt, poss[j+1], nbr, xt);
    float* xsub = wf(ws, (size_t)Ncur * OCt);
    run_mlp_bn(stream, ws, mlp_up_sub[j], xt, Ncur, OCt, OCt, xsub);
    float* base = wf(ws, (size_t)Nup * OCt);
    run_mlp_bn(stream, ws, mlp_up[j], xs[j], Nup, OCt, OCt, base);
    int* idx3 = wi(ws, (size_t)Nup * 3);
    float* d3 = wf(ws, (size_t)Nup * 3);
    run_knn(stream, poss[j], poss[j+1], idx3, d3, Nup, Ncur, 3, 0);
    interp_add_kernel<<<cdiv((size_t)Nup * OCt, 256), 256, 0, stream>>>(
        base, xsub, idx3, d3, xalt, Nup, OCt);
    ws.off = mark;
    float* t = xcur; xcur = xalt; xalt = t;
  }

  // ---- end transformer + classification head ----
  {
    size_t mark = ws.off;
    int* nbr = wi(ws, (size_t)8192 * 16);
    run_knn(stream, poss[0], poss[0], nbr, nullptr, 8192, 8192, 16, 1);
    float* xt = wf(ws, (size_t)8192 * 32);
    run_tblock(stream, ws, end_t, xcur, 8192, 32, 32, 32, poss[0], nbr, xt);
    float* h1 = wf(ws, (size_t)8192 * 64);
    gemm(stream, xt, head1.w, head1.b, h1, 8192, 64, 32, 1);
    float* h2 = wf(ws, (size_t)8192 * 64);
    gemm(stream, h1, head2.w, head2.b, h2, 8192, 64, 64, 1);
    float* logits = (float*)d_out;
    gemm(stream, h2, head3.w, head3.b, logits, 8192, 13, 64, 0);
    logsoftmax_kernel<<<cdiv(8192, 256), 256, 0, stream>>>(logits, 8192, 13);
    ws.off = mark;
  }
}
